// HybridClassifier_65481071404811
// MI455X (gfx1250) — compile-verified, matrix operand order checked by reading, not run on hardware
//
#include <hip/hip_runtime.h>
#include <hip/hip_bf16.h>

typedef __attribute__((ext_vector_type(16))) _Float16 v16h;
typedef __attribute__((ext_vector_type(8)))  float    v8f;
typedef __attribute__((ext_vector_type(2)))  float    v2f;

#define BLOCK_WAVES     8      // 256 threads, one 16-row tile per block
#define ROWS_PER_BLOCK  16
#define FEAT_K          832    // 13 patch-groups * 16 patches * 4 wires (zeros past 784)
#define H1_N            128    // fc1 out 120 padded
#define H2_N            96     // fc2 out 84 padded

__global__ __launch_bounds__(BLOCK_WAVES * 32)
void qcnn_fused(const float* __restrict__ x,  const float* __restrict__ U,
                const float* __restrict__ w1, const float* __restrict__ b1,
                const float* __restrict__ w2, const float* __restrict__ b2,
                const float* __restrict__ w3, const float* __restrict__ b3,
                float* __restrict__ out, int batch) {
  // One shared staging buffer per block (30.7 KB): feats[16][832] + h1[16][128]
  __shared__ _Float16 feats[ROWS_PER_BLOCK * FEAT_K];
  __shared__ _Float16 h1[ROWS_PER_BLOCK * H1_N];
  _Float16* h2 = feats;                       // alias: feats dead once fc1 is done

  const int wave    = threadIdx.x >> 5;
  const int lane    = threadIdx.x & 31;
  const int hf      = lane >> 4;              // 0: lanes 0-15, 1: lanes 16-31
  const int l16     = lane & 15;
  const int rowBase = blockIdx.x * ROWS_PER_BLOCK;

  __builtin_prefetch(w1, 0, 1);               // global_prefetch_b8: warm L2 for weights
  __builtin_prefetch(w2, 0, 1);

  // ===== Stage A: quantum filter.  psi' = U @ psi via V_WMMA_F32_16X16X4_F32 =====
  // A = U fragment (16x4 per k-chunk): lanes 0-15 hold K={4kc,4kc+1}, lanes 16-31 K={4kc+2,4kc+3}
  v2f ufrag[4];
  #pragma unroll
  for (int kc = 0; kc < 4; ++kc) {
    const int k = kc * 4 + hf * 2;
    ufrag[kc].x = U[l16 * 16 + k];
    ufrag[kc].y = U[l16 * 16 + k + 1];
  }

  // 208 (row, patch-group) tiles spread over 8 waves; unroll 2 so two
  // independent WMMA dependency chains co-execute within the wave.
  #pragma unroll 2
  for (int t = wave; t < ROWS_PER_BLOCK * 13; t += BLOCK_WAVES) {
    const int  r     = t & 15;
    const int  g     = t >> 4;
    const int  p     = g * 16 + l16;          // patch column (same for both lane halves)
    const bool valid = (p < 196);
    const int  pp    = valid ? p : 0;
    const int  pr    = pp / 14, pc = pp % 14;
    const float* xrow = x + (size_t)(rowBase + r) * 784;

    float c[4], s[4];
    #pragma unroll
    for (int d = 0; d < 4; ++d) {             // 2x2 patch wires
      const float t2 = 0.5f * xrow[(2 * pr + (d >> 1)) * 28 + (2 * pc + (d & 1))];
      c[d] = __cosf(t2);
      s[d] = __sinf(t2);
    }
    // psi_j = q01[j>>2] * q23[j&3]; this lane only needs q23 at indices {2*hf, 2*hf+1}
    float q01[4];
    q01[0] = c[0] * c[1]; q01[1] = c[0] * s[1];
    q01[2] = s[0] * c[1]; q01[3] = s[0] * s[1];
    float q23a = (hf ? s[2] : c[2]) * c[3];
    float q23b = (hf ? s[2] : c[2]) * s[3];
    if (!valid) { q23a = 0.f; q23b = 0.f; }   // zero pad columns -> feats pad = 0

    v8f acc = {0.f, 0.f, 0.f, 0.f, 0.f, 0.f, 0.f, 0.f};
    #pragma unroll
    for (int kc = 0; kc < 4; ++kc) {          // j0 = kc*4 + hf*2
      v2f bf;
      bf.x = q01[kc] * q23a;
      bf.y = q01[kc] * q23b;
      acc = __builtin_amdgcn_wmma_f32_16x16x4_f32(
          false, ufrag[kc], false, bf, (short)0, acc, false, false);
    }
    // PauliZ expectations: sum_i psi'^2 * (1 - 2*bit_w(i)); lane holds i = r8 + hf*8
    float mz[4] = {0.f, 0.f, 0.f, 0.f};
    #pragma unroll
    for (int r8 = 0; r8 < 8; ++r8) {
      const int   i  = r8 + hf * 8;
      const float v  = acc[r8];
      const float v2 = v * v;
      #pragma unroll
      for (int w = 0; w < 4; ++w)
        mz[w] += (((i >> (3 - w)) & 1) ? -v2 : v2);
    }
    #pragma unroll
    for (int w = 0; w < 4; ++w)
      mz[w] += __shfl_xor(mz[w], 16, 32);     // combine lane halves (same patch column)
    if (hf == 0) {
      _Float16* fr = feats + r * FEAT_K + p * 4;
      #pragma unroll
      for (int w = 0; w < 4; ++w) fr[w] = (_Float16)mz[w];
    }
  }

  __syncthreads();

  // ===== Stage B: MLP with V_WMMA_F32_16X16X32_F16 =====
  const int kb = hf * 8;                      // per-lane K base for 16-bit fragments

  // fc1: [16 x 832] @ [832 x 128]; each wave owns one N-tile (nt = wave).
  // K-loop: columns >= 800 are all-zero on both sides -> stop at ks=24;
  // ks 0..23 never touch k >= 784, ks == 24 needs the bounds guard.
  // Two independent accumulators halve the WMMA RAW dependency chain.
  {
    const int    nt     = wave;
    const int    n      = nt * 16 + l16;
    const bool   nvalid = (n < 120);
    const float* wrow   = w1 + (size_t)(nvalid ? n : 0) * 784;
    v8f acc0 = {0.f, 0.f, 0.f, 0.f, 0.f, 0.f, 0.f, 0.f};
    v8f acc1 = {0.f, 0.f, 0.f, 0.f, 0.f, 0.f, 0.f, 0.f};
    for (int ks = 0; ks < 24; ks += 2) {
      #pragma unroll
      for (int u = 0; u < 2; ++u) {
        const int k0 = (ks + u) * 32;
        const _Float16* fA = feats + l16 * FEAT_K + k0 + kb;
        const float*    fB = wrow + k0 + kb;
        v16h Af, Bf;
        #pragma unroll
        for (int e = 0; e < 8; ++e) {         // two contiguous 8-element segments
          Af[e]     = fA[e];
          Af[e + 8] = fA[e + 16];
          Bf[e]     = (_Float16)fB[e];
          Bf[e + 8] = (_Float16)fB[e + 16];
        }
        if (u == 0)
          acc0 = __builtin_amdgcn_wmma_f32_16x16x32_f16(
              false, Af, false, Bf, (short)0, acc0, false, false);
        else
          acc1 = __builtin_amdgcn_wmma_f32_16x16x32_f16(
              false, Af, false, Bf, (short)0, acc1, false, false);
      }
    }
    { // ks = 24: k in [768, 800), weights valid only below 784
      const int k0 = 768;
      const _Float16* fA = feats + l16 * FEAT_K + k0 + kb;
      v16h Af, Bf;
      #pragma unroll
      for (int e = 0; e < 16; ++e) {
        const int K  = kb + (e < 8 ? e : e + 8);
        const int kk = k0 + K;
        Af[e] = fA[(e < 8 ? e : e + 8)];
        Bf[e] = (nvalid && kk < 784) ? (_Float16)wrow[kk] : (_Float16)0.f;
      }
      acc0 = __builtin_amdgcn_wmma_f32_16x16x32_f16(
          false, Af, false, Bf, (short)0, acc0, false, false);
    }
    const v8f acc = acc0 + acc1;
    const float bias = nvalid ? b1[n] : 0.f;
    #pragma unroll
    for (int r8 = 0; r8 < 8; ++r8) {
      const int m = r8 + kb;                  // D row
      float v = acc[r8] + bias;
      v = v > 0.f ? v : 0.f;                  // relu
      h1[m * H1_N + n] = nvalid ? (_Float16)v : (_Float16)0.f;
    }
  }

  __syncthreads();                            // all of h1 visible; feats free to reuse

  // fc2: [16 x 128] @ [128 x 96]; waves 0..5 own one N-tile each.
  if (wave < 6) {
    const int    nt     = wave;
    const int    n      = nt * 16 + l16;
    const bool   nvalid = (n < 84);
    const float* wrow   = w2 + (size_t)(nvalid ? n : 0) * 120;
    v8f acc = {0.f, 0.f, 0.f, 0.f, 0.f, 0.f, 0.f, 0.f};
    #pragma unroll
    for (int ks = 0; ks < 3; ++ks) {          // k < 96 + 24 < 120: no guard needed
      const int k0 = ks * 32;
      const _Float16* fA = h1 + l16 * H1_N + k0 + kb;
      const float*    fB = wrow + k0 + kb;
      v16h Af, Bf;
      #pragma unroll
      for (int e = 0; e < 8; ++e) {
        Af[e]     = fA[e];
        Af[e + 8] = fA[e + 16];
        Bf[e]     = (_Float16)fB[e];
        Bf[e + 8] = (_Float16)fB[e + 16];
      }
      acc = __builtin_amdgcn_wmma_f32_16x16x32_f16(
          false, Af, false, Bf, (short)0, acc, false, false);
    }
    { // ks = 3: k in [96, 128), weights valid only below 120
      const int k0 = 96;
      const _Float16* fA = h1 + l16 * H1_N + k0 + kb;
      v16h Af, Bf;
      #pragma unroll
      for (int e = 0; e < 16; ++e) {
        const int K  = kb + (e < 8 ? e : e + 8);
        const int kk = k0 + K;
        Af[e] = fA[(e < 8 ? e : e + 8)];
        Bf[e] = (nvalid && kk < 120) ? (_Float16)wrow[kk] : (_Float16)0.f;
      }
      acc = __builtin_amdgcn_wmma_f32_16x16x32_f16(
          false, Af, false, Bf, (short)0, acc, false, false);
    }
    const float bias = nvalid ? b2[n] : 0.f;
    #pragma unroll
    for (int r8 = 0; r8 < 8; ++r8) {
      const int m = r8 + kb;
      float v = acc[r8] + bias;
      v = v > 0.f ? v : 0.f;
      h2[m * H2_N + n] = nvalid ? (_Float16)v : (_Float16)0.f;
    }
  }

  __syncthreads();

  // fc3 + analytic quantum head: wave 0, one batch row per lane (lanes 0-15)
  if (wave == 0 && hf == 0) {
    const int m = l16;
    float acc3 = b3[0];
    #pragma unroll 4
    for (int j = 0; j < 84; ++j)
      acc3 += (float)h2[m * H2_N + j] * w3[j];
    const float ev = 0.5f * (1.f + __sinf(acc3));   // H then RY: P(|1>)
    const float pv = 1.f / (1.f + __expf(-ev));     // sigmoid
    const int row  = rowBase + m;
    out[row]         = pv;
    out[batch + row] = 1.f - pv;
  }
}

extern "C" void kernel_launch(void* const* d_in, const int* in_sizes, int n_in,
                              void* d_out, int out_size, void* d_ws, size_t ws_size,
                              hipStream_t stream) {
  const float* x  = (const float*)d_in[0];
  const float* U  = (const float*)d_in[1];
  const float* w1 = (const float*)d_in[2];
  const float* b1 = (const float*)d_in[3];
  const float* w2 = (const float*)d_in[4];
  const float* b2 = (const float*)d_in[5];
  const float* w3 = (const float*)d_in[6];
  const float* b3 = (const float*)d_in[7];
  float* out = (float*)d_out;

  const int batch  = in_sizes[0] / 784;             // 8192
  const int blocks = batch / ROWS_PER_BLOCK;        // 512 blocks x 8 waves = 4096 waves
  qcnn_fused<<<blocks, BLOCK_WAVES * 32, 0, stream>>>(
      x, U, w1, b1, w2, b2, w3, b3, out, batch);
}
